// RNN_28355374088434
// MI455X (gfx1250) — matrix operand order repaired and checked by reading, not compiled, not run on hardware
//
#include <hip/hip_runtime.h>

typedef float  v4f   __attribute__((ext_vector_type(4)));
typedef float  v8f   __attribute__((ext_vector_type(8)));
typedef __bf16 v8bf  __attribute__((ext_vector_type(8)));
typedef __bf16 v16bf __attribute__((ext_vector_type(16)));

#define T_DIM 512
#define B_DIM 64
#define DIN   128
#define DH    1024
#define DOUT  256
#define RANK  64

// ---------------------------------------------------------------------------
// WMMA wrapper: bf16 operands, fp32 accumulate (the CDNA5 throughput path)
// ---------------------------------------------------------------------------
__device__ __forceinline__ v8f wmma32(v16bf a, v16bf b, v8f c) {
  return __builtin_amdgcn_wmma_f32_16x16x32_bf16(false, a, false, b, (short)0, c,
                                                 false, false);
}

// Split 8 consecutive fp32 into bf16 hi/lo at vector offset BASE.
template <int BASE>
__device__ __forceinline__ void split8(const float* __restrict__ p, v16bf& h, v16bf& l) {
  v4f f0 = *(const v4f*)p;
  v4f f1 = *(const v4f*)(p + 4);
#pragma unroll
  for (int i = 0; i < 4; ++i) {
    __bf16 h0 = (__bf16)f0[i];
    h[BASE + i] = h0;
    l[BASE + i] = (__bf16)(f0[i] - (float)h0);
    __bf16 h1 = (__bf16)f1[i];
    h[BASE + 4 + i] = h1;
    l[BASE + 4 + i] = (__bf16)(f1[i] - (float)h1);
  }
}

// Split the elementwise SUM of two fp32 streams (S0+S1 partials) into hi/lo.
template <int BASE>
__device__ __forceinline__ void split8sum(const float* __restrict__ p0,
                                          const float* __restrict__ p1,
                                          v16bf& h, v16bf& l) {
  v4f f0 = *(const v4f*)p0 + *(const v4f*)p1;
  v4f f1 = *(const v4f*)(p0 + 4) + *(const v4f*)(p1 + 4);
#pragma unroll
  for (int i = 0; i < 4; ++i) {
    __bf16 h0 = (__bf16)f0[i];
    h[BASE + i] = h0;
    l[BASE + i] = (__bf16)(f0[i] - (float)h0);
    __bf16 h1 = (__bf16)f1[i];
    h[BASE + 4 + i] = h1;
    l[BASE + 4 + i] = (__bf16)(f1[i] - (float)h1);
  }
}

// Gather a 16-bf16 A/B fragment from two contiguous 8-element chunks.
__device__ __forceinline__ v16bf frag2(const __bf16* __restrict__ c0,
                                       const __bf16* __restrict__ c1) {
  v8bf a = *(const v8bf*)c0;
  v8bf b = *(const v8bf*)c1;
  v16bf r;
#pragma unroll
  for (int i = 0; i < 8; ++i) { r[i] = a[i]; r[8 + i] = b[i]; }
  return r;
}

// ---------------------------------------------------------------------------
// Kernel 0: pre-split Wi[DH,DIN], Wo[DOUT,DH], V[RANK,DH], U[DH,RANK]
// into bf16 hi/lo arrays in d_ws (2 MB total, L2-resident).
// ---------------------------------------------------------------------------
__global__ __launch_bounds__(256) void cvt_weights(const float* __restrict__ Wi,
                                                   const float* __restrict__ Wo,
                                                   const float* __restrict__ V,
                                                   const float* __restrict__ U,
                                                   __bf16* __restrict__ wi_h, __bf16* __restrict__ wi_l,
                                                   __bf16* __restrict__ wo_h, __bf16* __restrict__ wo_l,
                                                   __bf16* __restrict__ v_h,  __bf16* __restrict__ v_l,
                                                   __bf16* __restrict__ u_h,  __bf16* __restrict__ u_l) {
  const int i = blockIdx.x * 256 + threadIdx.x;
  if (i < DH * DIN)  { float f = Wi[i]; __bf16 h = (__bf16)f; wi_h[i] = h; wi_l[i] = (__bf16)(f - (float)h); }
  if (i < DOUT * DH) { float f = Wo[i]; __bf16 h = (__bf16)f; wo_h[i] = h; wo_l[i] = (__bf16)(f - (float)h); }
  if (i < RANK * DH) { float f = V[i];  __bf16 h = (__bf16)f; v_h[i]  = h; v_l[i]  = (__bf16)(f - (float)h); }
  if (i < DH * RANK) { float f = U[i];  __bf16 h = (__bf16)f; u_h[i]  = h; u_l[i]  = (__bf16)(f - (float)h); }
}

// ---------------------------------------------------------------------------
// bf16x3 GEMM strip: wave computes 16(M) x 64(N) of C = A(fp32) * W^T
// (W pre-split bf16 [N,K] row-major).
// ---------------------------------------------------------------------------
template <int K, int LDC>
__device__ __forceinline__ void gemm_strip_bf16x3(const float* __restrict__ A,
                                                  const __bf16* __restrict__ Wh,
                                                  const __bf16* __restrict__ Wl,
                                                  float* __restrict__ C,
                                                  const float* __restrict__ bias,
                                                  int lane16, int hh) {
  v8f acc[4] = {};
  const float* Ar = A + (size_t)lane16 * K;
#pragma unroll 2
  for (int k = 0; k < K; k += 32) {
    v16bf ah, al;
    split8<0>(Ar + k + hh * 16, ah, al);
    split8<8>(Ar + k + hh * 16 + 8, ah, al);
#pragma unroll
    for (int j = 0; j < 4; ++j) {
      const size_t boff = (size_t)(j * 16 + lane16) * K + k + hh * 16;
      v16bf bh = *(const v16bf*)(Wh + boff);
      v16bf bl = *(const v16bf*)(Wl + boff);
      acc[j] = wmma32(ah, bh, acc[j]);
      acc[j] = wmma32(al, bh, acc[j]);
      acc[j] = wmma32(ah, bl, acc[j]);
    }
  }
#pragma unroll
  for (int j = 0; j < 4; ++j) {
    const float bj = bias ? bias[j * 16 + lane16] : 0.0f;
    float* cp = C + (size_t)(8 * hh) * LDC + j * 16 + lane16;
#pragma unroll
    for (int v = 0; v < 8; ++v) cp[(size_t)v * LDC] = acc[j][v] + bj;
  }
}

// ---------------------------------------------------------------------------
// Kernel 1: zi = x @ Wi^T  (M=32768, N=DH, K=DIN) -> hidden region of d_out
// ---------------------------------------------------------------------------
__global__ __launch_bounds__(256) void zi_gemm(const float* __restrict__ x,
                                               const __bf16* __restrict__ wi_h,
                                               const __bf16* __restrict__ wi_l,
                                               float* __restrict__ hid) {
  const int lane16 = threadIdx.x & 15;
  const int hh     = (threadIdx.x >> 4) & 1;
  const int wave   = threadIdx.x >> 5;
  const int wid    = blockIdx.x * 8 + wave;
  const int m0     = (wid / (DH / 64)) * 16;
  const int n0     = (wid % (DH / 64)) * 64;
  gemm_strip_bf16x3<DIN, DH>(x + (size_t)m0 * DIN,
                             wi_h + (size_t)n0 * DIN, wi_l + (size_t)n0 * DIN,
                             hid + (size_t)m0 * DH + n0, nullptr, lane16, hh);
}

// ---------------------------------------------------------------------------
// Kernel 2: sequential recurrence, ONE workgroup (32 wave32 waves).
// h kept in LDS as bf16 hi/lo (exact fp32 h goes to global hidden[t]).
// LDS: h_hi 128K + h_lo 128K + S0 16K + S1 16K = 288 KB <= 320 KB/WGP.
// ---------------------------------------------------------------------------
__global__ __launch_bounds__(1024) void recurrence(float* __restrict__ hid,   // zi in, hidden out
                                                   const __bf16* __restrict__ v_h,
                                                   const __bf16* __restrict__ v_l,
                                                   const __bf16* __restrict__ u_h,
                                                   const __bf16* __restrict__ u_l,
                                                   const float* __restrict__ bh) {
  extern __shared__ char lds_raw[];
  __bf16* hHi = (__bf16*)lds_raw;                       // [B_DIM][DH]
  __bf16* hLo = hHi + (size_t)B_DIM * DH;               // [B_DIM][DH]
  float*  S0  = (float*)(hLo + (size_t)B_DIM * DH);     // [B_DIM][RANK]
  float*  S1  = S0 + B_DIM * RANK;                      // [B_DIM][RANK]

  const int tid    = threadIdx.x;
  const int lane16 = tid & 15;
  const int hh     = (tid >> 4) & 1;
  const int wave   = tid >> 5;                          // 0..31

  for (int i = tid; i < B_DIM * DH; i += 1024) { hHi[i] = (__bf16)0.0f; hLo[i] = (__bf16)0.0f; }
  __syncthreads();

  const int p1t   = wave >> 1;                          // 0..15 (S tile)
  const int khalf = wave & 1;                           // K half
  const int bm1   = (p1t >> 2) * 16;
  const int rn1   = (p1t & 3) * 16;
  float* Sbuf     = khalf ? S1 : S0;
  const int k0    = khalf * (DH / 2);

  for (int t = 0; t < T_DIM; ++t) {
    // ---- Phase 1: partial S = h @ V^T over this wave's K-half (bf16x3) ----
    {
      const __bf16* Ah = hHi + (size_t)(bm1 + lane16) * DH;
      const __bf16* Al = hLo + (size_t)(bm1 + lane16) * DH;
      const __bf16* Bh = v_h + (size_t)(rn1 + lane16) * DH;
      const __bf16* Bl = v_l + (size_t)(rn1 + lane16) * DH;
      v8f acc = {};
#pragma unroll 4
      for (int k = k0; k < k0 + DH / 2; k += 32) {
        v16bf ah = frag2(Ah + k + hh * 16, Ah + k + hh * 16 + 8);   // ds_load
        v16bf al = frag2(Al + k + hh * 16, Al + k + hh * 16 + 8);   // ds_load
        v16bf bh_ = *(const v16bf*)(Bh + k + hh * 16);             // global (L2-hot)
        v16bf bl_ = *(const v16bf*)(Bl + k + hh * 16);
        acc = wmma32(ah, bh_, acc);
        acc = wmma32(al, bh_, acc);
        acc = wmma32(ah, bl_, acc);
      }
      float* sp = Sbuf + (size_t)(bm1 + 8 * hh) * RANK + rn1 + lane16;
#pragma unroll
      for (int v = 0; v < 8; ++v) sp[(size_t)v * RANK] = acc[v];
    }
    __syncthreads();

    // ---- Phase 2: h_new = relu((S0+S1) @ U^T + bh + zi_t) ----
    float* zt = hid + (size_t)t * B_DIM * DH;
    for (int tt = wave; tt < (B_DIM / 16) * (DH / 16); tt += 32) {  // 8 tiles/wave
      const int b0 = (tt >> 6) * 16;
      const int j0 = (tt & 63) * 16;
      const float* A0 = S0 + (size_t)(b0 + lane16) * RANK;
      const float* A1 = S1 + (size_t)(b0 + lane16) * RANK;
      const __bf16* Bh = u_h + (size_t)(j0 + lane16) * RANK;
      const __bf16* Bl = u_l + (size_t)(j0 + lane16) * RANK;
      v8f acc = {};
#pragma unroll
      for (int k = 0; k < RANK; k += 32) {
        v16bf ah, al;
        split8sum<0>(A0 + k + hh * 16, A1 + k + hh * 16, ah, al);
        split8sum<8>(A0 + k + hh * 16 + 8, A1 + k + hh * 16 + 8, ah, al);
        v16bf bh_ = *(const v16bf*)(Bh + k + hh * 16);
        v16bf bl_ = *(const v16bf*)(Bl + k + hh * 16);
        acc = wmma32(ah, bh_, acc);
        acc = wmma32(al, bh_, acc);
        acc = wmma32(ah, bl_, acc);
      }
#pragma unroll
      for (int v = 0; v < 8; ++v) {
        const int b = b0 + v + 8 * hh;
        const int j = j0 + lane16;
        float val = acc[v] + bh[j] + zt[(size_t)b * DH + j];
        val = fmaxf(val, 0.0f);
        zt[(size_t)b * DH + j] = val;            // exact fp32 hidden[t] (in-place)
        __bf16 hb = (__bf16)val;                 // bf16 hi/lo state for next step
        hHi[(size_t)b * DH + j] = hb;
        hLo[(size_t)b * DH + j] = (__bf16)(val - (float)hb);
      }
    }
    __syncthreads();
  }
}

// ---------------------------------------------------------------------------
// Kernel 3: out = hidden @ Wo^T + bo  (M=32768, N=DOUT, K=DH)
// ---------------------------------------------------------------------------
__global__ __launch_bounds__(256) void out_gemm(const float* __restrict__ hid,
                                                const __bf16* __restrict__ wo_h,
                                                const __bf16* __restrict__ wo_l,
                                                const float* __restrict__ bo,
                                                float* __restrict__ out) {
  const int lane16 = threadIdx.x & 15;
  const int hh     = (threadIdx.x >> 4) & 1;
  const int wave   = threadIdx.x >> 5;
  const int wid    = blockIdx.x * 8 + wave;
  const int m0     = (wid / (DOUT / 64)) * 16;
  const int n0     = (wid % (DOUT / 64)) * 64;
  gemm_strip_bf16x3<DH, DOUT>(hid + (size_t)m0 * DH,
                              wo_h + (size_t)n0 * DH, wo_l + (size_t)n0 * DH,
                              out + (size_t)m0 * DOUT + n0, bo + n0, lane16, hh);
}

// ---------------------------------------------------------------------------
extern "C" void kernel_launch(void* const* d_in, const int* in_sizes, int n_in,
                              void* d_out, int out_size, void* d_ws, size_t ws_size,
                              hipStream_t stream) {
  const float* x  = (const float*)d_in[0];
  const float* Wi = (const float*)d_in[1];
  const float* U  = (const float*)d_in[2];
  const float* V  = (const float*)d_in[3];
  const float* bh = (const float*)d_in[4];
  const float* Wo = (const float*)d_in[5];
  const float* bo = (const float*)d_in[6];

  float* hid = (float*)d_out;                         // [T*B, DH] (zi -> hidden)
  float* out = hid + (size_t)T_DIM * B_DIM * DH;      // [T*B, DOUT]

  // Workspace layout: bf16 hi/lo copies of Wi, Wo, V, U (2 MB total).
  __bf16* wi_h = (__bf16*)d_ws;
  __bf16* wi_l = wi_h + (size_t)DH * DIN;
  __bf16* wo_h = wi_l + (size_t)DH * DIN;
  __bf16* wo_l = wo_h + (size_t)DOUT * DH;
  __bf16* v_h  = wo_l + (size_t)DOUT * DH;
  __bf16* v_l  = v_h  + (size_t)RANK * DH;
  __bf16* u_h  = v_l  + (size_t)RANK * DH;
  __bf16* u_l  = u_h  + (size_t)DH * RANK;

  const size_t lds_bytes = 2 * (size_t)B_DIM * DH * sizeof(__bf16)    // h hi+lo
                         + 2 * (size_t)B_DIM * RANK * sizeof(float);  // S0,S1  => 288 KB
  static bool attr_set = false;
  if (!attr_set) {
    (void)hipFuncSetAttribute((const void*)recurrence,
                              hipFuncAttributeMaxDynamicSharedMemorySize,
                              (int)lds_bytes);
    attr_set = true;
  }

  // 0) split weights into bf16 hi/lo
  cvt_weights<<<(DOUT * DH + 255) / 256, 256, 0, stream>>>(
      Wi, Wo, V, U, wi_h, wi_l, wo_h, wo_l, v_h, v_l, u_h, u_l);
  // 1) zi = x @ Wi^T -> hidden region (bf16x3 WMMA)
  zi_gemm<<<(T_DIM * B_DIM / 16) * (DH / 64) / 8, 256, 0, stream>>>(x, wi_h, wi_l, hid);
  // 2) sequential recurrence (bf16x3 WMMA, LDS-resident hi/lo state), single WGP
  recurrence<<<1, 1024, lds_bytes, stream>>>(hid, v_h, v_l, u_h, u_l, bh);
  // 3) output = hidden @ Wo^T + bo (bf16x3 WMMA)
  out_gemm<<<(T_DIM * B_DIM / 16) * (DOUT / 64) / 8, 256, 0, stream>>>(hid, wo_h, wo_l, bo, out);
}
